// RnnSender_15126874817269
// MI455X (gfx1250) — matrix operand order repaired and checked by reading, not compiled, not run on hardware
//
#include <hip/hip_runtime.h>
#include <hip/hip_bf16.h>

typedef __bf16 bf16_t;
typedef __attribute__((ext_vector_type(16))) __bf16 v16bf;
typedef __attribute__((ext_vector_type(8)))  __bf16 v8bf;
typedef __attribute__((ext_vector_type(8)))  float  v8f;

constexpr int Bd = 32, Td = 64, Vd = 32000, Ed = 512, Hd = 1024, Dd = 1024, H4 = 4096;
constexpr int BOSt = 1, EOSt = 3, PADt = 0;

// ---------------- f32 -> bf16 convert ----------------
__global__ void k_f32_to_bf16(const float* __restrict__ s, bf16_t* __restrict__ d, int n) {
    int i = blockIdx.x * blockDim.x + threadIdx.x;
    if (i < n) d[i] = (bf16_t)s[i];
}

// ---------------- bf16 WMMA GEMM: C[32,N] = A[32,K] * W[N,K]^T (+bias) ----------------
// One wave computes a full-M 32x64 tile: 2 M-tiles x 4 N-subtiles = 8
// v_wmma_f32_16x16x32_bf16 per K-step; B fragments shared across both M-tiles
// (halves the dominant weight stream). The K-loop is unrolled 2x into a true
// ping-pong over two fragment buffer sets: loads for the next 32-wide K-step go
// to the idle buffer while WMMAs consume the other, so no register copies and
// only partial s_wait_loadcnt before each WMMA group. Requires K % 64 == 0.
// EXEC stays all-ones (early-out is wave-uniform).
__global__ void k_gemm_bf16_wmma(const bf16_t* __restrict__ A,
                                 const bf16_t* __restrict__ W,
                                 const float* __restrict__ bias0,
                                 const float* __restrict__ bias1,
                                 float* __restrict__ C,
                                 int N, int K, int accumulate)
{
    const int lane  = threadIdx.x & 31;
    const int gwave = blockIdx.x * (blockDim.x >> 5) + (threadIdx.x >> 5);
    const int total = N >> 6;                 // one wave per 64-wide N tile, full M
    if (gwave >= total) return;               // wave-uniform: EXEC stays full otherwise

    const int n0   = gwave << 6;
    const int half = lane >> 4;               // 0: lanes 0-15, 1: lanes 16-31
    const int l15  = lane & 15;

    v8f acc[2][4];
#pragma unroll
    for (int mt = 0; mt < 2; ++mt) {
#pragma unroll
        for (int s = 0; s < 4; ++s) {
            const int n = n0 + s * 16 + l15;
            if (accumulate) {
#pragma unroll
                for (int r = 0; r < 8; ++r)
                    acc[mt][s][r] = C[(size_t)(mt * 16 + half * 8 + r) * N + n];
            } else {
                float base = 0.f;
                if (bias0) base += bias0[n];
                if (bias1) base += bias1[n];
#pragma unroll
                for (int r = 0; r < 8; ++r) acc[mt][s][r] = base;
            }
        }
    }

    // A fragment: lane half 0 -> K = {k..k+7, k+16..k+23}; half 1 -> {k+8..k+15, k+24..k+31}
    const bf16_t* aRow0 = A + (size_t)(l15)      * K + half * 8;  // M-tile 0 (rows 0-15)
    const bf16_t* aRow1 = A + (size_t)(16 + l15) * K + half * 8;  // M-tile 1 (rows 16-31)
    // B fragment: lane half 0 -> K = {k..k+7, k+16..k+23}; half 1 -> {k+8..k+15, k+24..k+31}
    const bf16_t* wBase = W + (size_t)(n0 + l15) * K + half * 8;

    auto loadA = [&](const bf16_t* row, int k) -> v16bf {
        v8bf lo = *(const v8bf*)(row + k);
        v8bf hi = *(const v8bf*)(row + k + 16);
        return __builtin_shufflevector(lo, hi, 0,1,2,3,4,5,6,7,8,9,10,11,12,13,14,15);
    };
    auto loadB = [&](int s, int k) -> v16bf {
        const bf16_t* wp = wBase + (size_t)(s * 16) * K + k;
        v8bf lo = *(const v8bf*)wp;
        v8bf hi = *(const v8bf*)(wp + 16);
        return __builtin_shufflevector(lo, hi, 0,1,2,3,4,5,6,7,8,9,10,11,12,13,14,15);
    };

    v16bf aF[2][2];   // [buffer][m-tile]
    v16bf bF[2][4];   // [buffer][n-subtile]

    auto loadFrag = [&](int buf, int k) {
        aF[buf][0] = loadA(aRow0, k);
        aF[buf][1] = loadA(aRow1, k);
#pragma unroll
        for (int s = 0; s < 4; ++s) bF[buf][s] = loadB(s, k);
    };
    auto domm = [&](int buf) {
#pragma unroll
        for (int s = 0; s < 4; ++s)
            acc[0][s] = __builtin_amdgcn_wmma_f32_16x16x32_bf16(false, aF[buf][0], false, bF[buf][s],
                                                                (short)0, acc[0][s], false, false);
#pragma unroll
        for (int s = 0; s < 4; ++s)
            acc[1][s] = __builtin_amdgcn_wmma_f32_16x16x32_bf16(false, aF[buf][1], false, bF[buf][s],
                                                                (short)0, acc[1][s], false, false);
    };

    // ping-pong pipeline over 64-wide K super-steps (K % 64 == 0)
    loadFrag(0, 0);
    int k = 0;
    for (; k + 64 < K; k += 64) {
        loadFrag(1, k + 32);
        domm(0);
        loadFrag(0, k + 64);
        domm(1);
    }
    loadFrag(1, k + 32);
    domm(0);
    domm(1);

#pragma unroll
    for (int mt = 0; mt < 2; ++mt) {
#pragma unroll
        for (int s = 0; s < 4; ++s) {
            float* cp = C + (size_t)(mt * 16 + half * 8) * N + n0 + s * 16 + l15;
#pragma unroll
            for (int r = 0; r < 8; ++r) cp[(size_t)r * N] = acc[mt][s][r];
        }
    }
}

// ---------------- gumbel + argmax over V, then gather x = emb[label] ----------------
__global__ void k_gumbel_argmax_gather(const float* __restrict__ logits, // [B,V]
                                       const float* __restrict__ u,      // [B,V] slice t
                                       const float* __restrict__ emb,    // [V,E] f32
                                       bf16_t* __restrict__ x,           // [B,E]
                                       int* __restrict__ labels,         // [B,T]
                                       int t)
{
    __shared__ float sval[256];
    __shared__ int   sidx[256];
    const int b = blockIdx.x, tid = threadIdx.x;
    float best = -3.4e38f; int bidx = 0x7fffffff;
    for (int v = tid; v < Vd; v += 256) {
        float uu = u[(size_t)b * Vd + v];
        float g  = -__logf(-__logf(uu + 1e-10f) + 1e-10f);
        float sc = logits[(size_t)b * Vd + v] + g;         // TAU = 1
        if (sc > best || (sc == best && v < bidx)) { best = sc; bidx = v; }
    }
    sval[tid] = best; sidx[tid] = bidx;
    __syncthreads();
    for (int s = 128; s > 0; s >>= 1) {
        if (tid < s) {
            if (sval[tid + s] > sval[tid] ||
                (sval[tid + s] == sval[tid] && sidx[tid + s] < sidx[tid])) {
                sval[tid] = sval[tid + s]; sidx[tid] = sidx[tid + s];
            }
        }
        __syncthreads();
    }
    const int lab = sidx[0];
    if (tid == 0) labels[b * Td + t] = lab;
    for (int e = tid; e < Ed; e += 256)
        x[b * Ed + e] = (bf16_t)emb[(size_t)lab * Ed + e];
}

// ---------------- LSTM cell elementwise ----------------
__device__ __forceinline__ float sigmf(float v) { return 1.f / (1.f + __expf(-v)); }

__global__ void k_lstm_cell(const float* __restrict__ gates, float* __restrict__ c,
                            bf16_t* __restrict__ hbf)
{
    int idx = blockIdx.x * blockDim.x + threadIdx.x;
    if (idx >= Bd * Hd) return;
    int b = idx / Hd, j = idx - b * Hd;
    const float* gr = gates + (size_t)b * H4;
    float gi = gr[j], gf = gr[Hd + j], gg = gr[2 * Hd + j], go = gr[3 * Hd + j];
    float cn = sigmf(gf) * c[idx] + sigmf(gi) * tanhf(gg);
    c[idx]   = cn;
    hbf[idx] = (bf16_t)(sigmf(go) * tanhf(cn));
}

// ---------------- h0 -> bf16, c = 0, x0 = emb[BOS] ----------------
__global__ void k_prep0(const float* __restrict__ h0, const float* __restrict__ emb,
                        float* __restrict__ c, bf16_t* __restrict__ hbf,
                        bf16_t* __restrict__ x)
{
    int idx = blockIdx.x * blockDim.x + threadIdx.x;
    if (idx < Bd * Hd) { c[idx] = 0.f; hbf[idx] = (bf16_t)h0[idx]; }
    if (idx < Bd * Ed) x[idx] = (bf16_t)emb[(size_t)BOSt * Ed + (idx % Ed)];
}

// ---------------- EOS masking / ids / lengths / per-row one-hot selector ----------------
__global__ void k_finalize_meta(int* __restrict__ labels, int* __restrict__ sel,
                                float* __restrict__ out_ids, float* __restrict__ out_len)
{
    int b = threadIdx.x;
    if (b >= Bd) return;
    labels[b * Td + Td - 1] = EOSt;
    int seen = 0, cnt = 0;
    for (int t = 0; t < Td; ++t) {
        int lab = labels[b * Td + t];
        seen += (lab == EOSt);
        int pad = (seen == 0);                 // pad_g: no EOS in [0..t] inclusive
        out_ids[b * Td + t] = (float)(pad ? lab : 0);
        sel[b * Td + t]     = pad ? lab : PADt;
        cnt += pad;
    }
    out_len[b] = (float)(cnt + 1);
}

// ---------------- message_logits: one-hot rows (forward value of straight-through) ----
__global__ void k_fill_logits(const int* __restrict__ sel, float* __restrict__ out_logits)
{
    int row = blockIdx.x;                      // b*T + t
    int s   = sel[row];
    float* dst = out_logits + (size_t)row * Vd;
    for (int v = threadIdx.x; v < Vd; v += blockDim.x)
        __builtin_nontemporal_store((v == s) ? 1.f : 0.f, dst + v);
}

extern "C" void kernel_launch(void* const* d_in, const int* in_sizes, int n_in,
                              void* d_out, int out_size, void* d_ws, size_t ws_size,
                              hipStream_t stream)
{
    (void)in_sizes; (void)n_in; (void)out_size; (void)ws_size;
    const float* image_hidden = (const float*)d_in[0];
    const float* gumbel_u     = (const float*)d_in[1];   // [T,B,V]
    const float* W_proj = (const float*)d_in[2];
    const float* b_proj = (const float*)d_in[3];
    const float* emb    = (const float*)d_in[4];         // stays f32 (gather only)
    const float* W_ih   = (const float*)d_in[5];
    const float* W_hh   = (const float*)d_in[6];
    const float* b_ih   = (const float*)d_in[7];
    const float* b_hh   = (const float*)d_in[8];
    const float* W_hv   = (const float*)d_in[9];
    const float* b_hv   = (const float*)d_in[10];

    // ---- workspace carve (256B aligned) ----
    char* p = (char*)d_ws;
    auto carve = [&](size_t bytes) -> char* {
        char* r = p; p += (bytes + 255) & ~(size_t)255; return r;
    };
    bf16_t* Wproj_bf = (bf16_t*)carve((size_t)Hd * Dd * 2);
    bf16_t* Wih_bf   = (bf16_t*)carve((size_t)H4 * Ed * 2);
    bf16_t* Whh_bf   = (bf16_t*)carve((size_t)H4 * Hd * 2);
    bf16_t* Whv_bf   = (bf16_t*)carve((size_t)Vd * Hd * 2);
    bf16_t* img_bf   = (bf16_t*)carve((size_t)Bd * Dd * 2);
    float*  h0f      = (float*) carve((size_t)Bd * Hd * 4);
    bf16_t* x_bf     = (bf16_t*)carve((size_t)Bd * Ed * 2);
    bf16_t* h_bf     = (bf16_t*)carve((size_t)Bd * Hd * 2);
    float*  c_f      = (float*) carve((size_t)Bd * Hd * 4);
    float*  gates_f  = (float*) carve((size_t)Bd * H4 * 4);
    float*  logits_f = (float*) carve((size_t)Bd * Vd * 4);
    int*    labels   = (int*)   carve((size_t)Bd * Td * 4);
    int*    sel      = (int*)   carve((size_t)Bd * Td * 4);

    float* out_ids    = (float*)d_out;                       // [B,T]
    float* out_logits = out_ids + (size_t)Bd * Td;           // [B,T,V]
    float* out_len    = out_logits + (size_t)Bd * Td * Vd;   // [B]

    auto cvt = [&](const float* s, bf16_t* d, int n) {
        k_f32_to_bf16<<<(n + 255) / 256, 256, 0, stream>>>(s, d, n);
    };
    auto gemm = [&](const bf16_t* A, const bf16_t* W, const float* bias0,
                    const float* bias1, float* C, int N, int K, int acc) {
        int waves  = N / 64;                                  // full-M 32x64 tile per wave
        int blocks = (waves + 3) / 4;                         // 4 waves / 128-thread block
        k_gemm_bf16_wmma<<<blocks, 128, 0, stream>>>(A, W, bias0, bias1, C, N, K, acc);
    };

    // ---- weight conversion (once per launch) ----
    cvt(W_proj, Wproj_bf, Hd * Dd);
    cvt(W_ih,   Wih_bf,   H4 * Ed);
    cvt(W_hh,   Whh_bf,   H4 * Hd);
    cvt(W_hv,   Whv_bf,   Vd * Hd);
    cvt(image_hidden, img_bf, Bd * Dd);

    // ---- h0 = image_hidden @ W_proj^T + b_proj ; c0 = 0 ; x0 = emb[BOS] ----
    gemm(img_bf, Wproj_bf, b_proj, nullptr, h0f, Hd, Dd, 0);
    k_prep0<<<(Bd * Hd + 255) / 256, 256, 0, stream>>>(h0f, emb, c_f, h_bf, x_bf);

    // ---- initial LSTM step ----
    gemm(x_bf, Wih_bf, b_ih, b_hh, gates_f, H4, Ed, 0);
    gemm(h_bf, Whh_bf, nullptr, nullptr, gates_f, H4, Hd, 1);
    k_lstm_cell<<<(Bd * Hd + 255) / 256, 256, 0, stream>>>(gates_f, c_f, h_bf);

    // ---- T decode steps (serial) ----
    for (int t = 0; t < Td; ++t) {
        gemm(h_bf, Whv_bf, b_hv, nullptr, logits_f, Vd, Hd, 0);
        const float* u_t = gumbel_u + (size_t)t * Bd * Vd;
        k_gumbel_argmax_gather<<<Bd, 256, 0, stream>>>(logits_f, u_t, emb, x_bf, labels, t);
        if (t < Td - 1) {                                   // last h/c unused
            gemm(x_bf, Wih_bf, b_ih, b_hh, gates_f, H4, Ed, 0);
            gemm(h_bf, Whh_bf, nullptr, nullptr, gates_f, H4, Hd, 1);
            k_lstm_cell<<<(Bd * Hd + 255) / 256, 256, 0, stream>>>(gates_f, c_f, h_bf);
        }
    }

    // ---- outputs ----
    k_finalize_meta<<<1, 32, 0, stream>>>(labels, sel, out_ids, out_len);
    k_fill_logits<<<Bd * Td, 256, 0, stream>>>(sel, out_logits);
}
